// MANNBaseEncoder_69578470195274
// MI455X (gfx1250) — matrix operand order repaired and checked by reading, not compiled, not run on hardware
//
#include <hip/hip_runtime.h>
#include <hip/hip_bf16.h>
#include <math.h>

#define IDIM   256
#define CDIM   512
#define NSLOT  128
#define MDIM   64
#define SEQ    128
#define BSZ    64
#define KCAT   832          // IDIM + MDIM + CDIM
#define KT_G   26           // 832/32
#define NT_G   128          // 2048/16
#define KT_H   16           // 512/32
#define KT_MLP 18           // 576/32
#define NT_MLP 32           // 512/16
#define THREADS 1024
#define OS_SZ  (SEQ*BSZ*(CDIM+MDIM))   // 128*64*576

typedef __attribute__((ext_vector_type(16))) __bf16        v16bf;
typedef __attribute__((ext_vector_type(8)))  float         v8f;
typedef __attribute__((ext_vector_type(16))) unsigned short v16u;

__device__ __forceinline__ unsigned short f2bf(float f) {
  unsigned int u = __builtin_bit_cast(unsigned int, f);
  u += 0x7fffu + ((u >> 16) & 1u);              // round-to-nearest-even
  return (unsigned short)(u >> 16);
}
// Fast transcendental path: v_exp_f32 / v_rcp_f32 (TRANS32, co-issues with VALU)
__device__ __forceinline__ float frcp(float x)  { return __builtin_amdgcn_rcpf(x); }
__device__ __forceinline__ float sigm(float x)  { return frcp(1.0f + __expf(-x)); }
__device__ __forceinline__ float ftanh(float x) { return 1.0f - 2.0f * frcp(1.0f + __expf(2.0f * x)); }

__device__ __forceinline__ v8f wmma_bf16(v16u a_, v16u b_, v8f c) {
  v16bf a = __builtin_bit_cast(v16bf, a_);
  v16bf b = __builtin_bit_cast(v16bf, b_);
  return __builtin_amdgcn_wmma_f32_16x16x32_bf16(false, a, false, b, (short)0, c,
                                                 false, false);
}
__device__ __forceinline__ v8f v8_splat(float x) {
  v8f r;
#pragma unroll
  for (int i = 0; i < 8; i++) r[i] = x;
  return r;
}

// A fragment (this file's contiguous k order, same as B packing): lane m=lane&15; low half holds K=[0..15],
// high half K=[16..31]. rowk0 points at element K=0 of this lane's row.
__device__ __forceinline__ v16u load_a(const unsigned short* rowk0, int hl) {
  v16u r;
  const unsigned short* p0 = rowk0 + hl * 16;
#pragma unroll
  for (int i = 0; i < 16; i++) { r[i] = p0[i]; }
  return r;
}

// ---------------------------------------------------------------- prep kernels
// Gate weights, fragment-swizzled: [kk(26)][ntile(128)][lane(32)][i(16)] bf16
__global__ void prep_wg(const float* __restrict__ Wih, const float* __restrict__ Whh,
                        unsigned short* __restrict__ dst) {
  int idx = blockIdx.x * blockDim.x + threadIdx.x;
  if (idx >= (26 << 16)) return;
  int i = idx & 15, lane = (idx >> 4) & 31, nt = (idx >> 9) & 127, kk = idx >> 16;
  int k = kk * 32 + (lane >> 4) * 16 + i;       // K index in concat(x) space
  int n = nt * 16 + (lane & 15);                // gate-output index 0..2047
  float v = (k < IDIM + MDIM) ? Wih[n * (IDIM + MDIM) + k]
                              : Whh[n * CDIM + (k - (IDIM + MDIM))];
  dst[idx] = f2bf(v);
}
// Head weights (kr,kw,e,a): [mat(4)][kk(16)][nt(4)][lane(32)][i(16)]
__global__ void prep_wh(const float* __restrict__ Wkr, const float* __restrict__ Wkw,
                        const float* __restrict__ We,  const float* __restrict__ Wa,
                        unsigned short* __restrict__ dst) {
  int idx = blockIdx.x * blockDim.x + threadIdx.x;
  if (idx >= (4 << 15)) return;
  int i = idx & 15, lane = (idx >> 4) & 31, nt = (idx >> 9) & 3;
  int kk = (idx >> 11) & 15, mat = idx >> 15;
  int k = kk * 32 + (lane >> 4) * 16 + i;
  int n = nt * 16 + (lane & 15);
  const float* W = (mat == 0) ? Wkr : (mat == 1) ? Wkw : (mat == 2) ? We : Wa;
  dst[idx] = f2bf(W[k * MDIM + n]);
}
// MLP weights: [kk(18)][nt(32)][lane(32)][i(16)]
__global__ void prep_wmlp(const float* __restrict__ Wm, unsigned short* __restrict__ dst) {
  int idx = blockIdx.x * blockDim.x + threadIdx.x;
  if (idx >= (18 << 14)) return;
  int i = idx & 15, lane = (idx >> 4) & 31, nt = (idx >> 9) & 31, kk = idx >> 14;
  int k = kk * 32 + (lane >> 4) * 16 + i;
  int n = nt * 16 + (lane & 15);
  dst[idx] = f2bf(Wm[k * CDIM + n]);
}
// Broadcast initial state into workspace
__global__ void prep_state(const float* __restrict__ mem0, const float* __restrict__ h0,
                           const float* __restrict__ c0, float* __restrict__ memW,
                           float* __restrict__ hW, float* __restrict__ cW) {
  int idx = blockIdx.x * blockDim.x + threadIdx.x;
  if (idx < BSZ * NSLOT * MDIM) memW[idx] = mem0[idx & (NSLOT * MDIM - 1)];
  if (idx < BSZ * CDIM) { hW[idx] = h0[idx & (CDIM - 1)]; cW[idx] = c0[idx & (CDIM - 1)]; }
}

// ---------------------------------------------------------------- main kernel
struct Params {
  const float* embs; const int* lens;
  const float* b_ih; const float* b_hh;
  const float* h0;   const float* c0; const float* r0;
  const float* b_kr; const float* b_kw; const float* b_e; const float* b_a;
  const float* b_mlp;
  const unsigned short* WgF; const unsigned short* WhF; const unsigned short* WmF;
  float* memW; float* hW; float* cW;
  float* out;
};

__global__ __launch_bounds__(THREADS, 1)
void mann_persistent(Params P) {
  // LDS-resident recurrent state (272 KB of the WGP's 320 KB)
  __shared__ __align__(16) unsigned short xcat[BSZ][840];   // [emb(256)|r(64)|h(512)] bf16, padded
  __shared__ __align__(16) unsigned short co2s[BSZ][520];   // co after MLP, bf16, padded
  __shared__ float keyb[4][BSZ][MDIM];                      // kr, kw, e(sigmoid), a
  __shared__ float sc[BSZ][NSLOT];                          // attention scores / weights

  const int tid  = threadIdx.x;
  const int lane = tid & 31;
  const int wave = tid >> 5;
  const int hl   = lane >> 4;      // lane half (K split of A/B fragments)
  const int lm   = lane & 15;      // M/N index within tile

  // ---- preamble: seed r and h slices of x_cat (bf16) -----------------------
  for (int idx = tid; idx < BSZ * MDIM; idx += THREADS) {
    int b = idx >> 6, m = idx & 63;
    xcat[b][IDIM + m] = f2bf(P.r0[m]);
  }
  for (int idx = tid; idx < BSZ * CDIM; idx += THREADS) {
    int b = idx >> 9, j = idx & 511;
    xcat[b][IDIM + MDIM + j] = f2bf(P.h0[j]);
  }
  __syncthreads();

  // ---- helper lambdas ------------------------------------------------------
  auto scores_pass = [&](const float (*key)[MDIM]) {   // s = mem·k / sqrt(M)
#pragma unroll 2
    for (int p = 0; p < 8; p++) {
      int idx = tid + p * THREADS; int b = idx >> 7, n = idx & 127;
      const float* mrow = P.memW + (size_t)(b * NSLOT + n) * MDIM;
      const float* kb = key[b];
      float s = 0.f;
#pragma unroll 8
      for (int m = 0; m < MDIM; m++) s += mrow[m] * kb[m];
      sc[b][n] = s * 0.125f;                          // 1/sqrt(64)
    }
  };
  auto softmax_pass = [&]() {
    if (tid < BSZ) {
      int b = tid; float mx = -1e30f;
      for (int n = 0; n < NSLOT; n++) mx = fmaxf(mx, sc[b][n]);
      float s = 0.f;
      for (int n = 0; n < NSLOT; n++) { float e = __expf(sc[b][n] - mx); sc[b][n] = e; s += e; }
      float inv = frcp(s);
      for (int n = 0; n < NSLOT; n++) sc[b][n] *= inv;
    }
  };
  auto mem_update = [&]() {                            // mem = mem*(1-w e) + w a
#pragma unroll 2
    for (int p = 0; p < 8; p++) {
      int idx = tid + p * THREADS; int b = idx >> 7, n = idx & 127;
      float w = sc[b][n];
      float* mrow = P.memW + (size_t)(b * NSLOT + n) * MDIM;
      const float* eb = keyb[2][b]; const float* ab = keyb[3][b];
#pragma unroll 8
      for (int m = 0; m < MDIM; m++)
        mrow[m] = mrow[m] * (1.0f - w * eb[m]) + w * ab[m];
    }
  };
  auto read_pass = [&](int t, bool toOut) {            // r = w·mem  -> xcat (+out)
#pragma unroll 2
    for (int p = 0; p < 4; p++) {
      int idx = tid + p * THREADS; int b = idx >> 6, m = idx & 63;
      const float* mcol = P.memW + (size_t)b * NSLOT * MDIM + m;
      float s = 0.f;
      for (int n = 0; n < NSLOT; n++) s += sc[b][n] * mcol[n * MDIM];
      xcat[b][IDIM + m] = f2bf(s);
      if (toOut) P.out[((size_t)t * BSZ + b) * (CDIM + MDIM) + CDIM + m] = s;
    }
  };
  // head projections: 64 tile jobs (4 mats x 4 ntiles x 4 mtiles), 2 per wave
  auto heads_pass = [&](const unsigned short* src, int srcStride) {
    for (int q = 0; q < 2; q++) {
      int job = wave * 2 + q;
      int mat = job >> 4, nt = (job >> 2) & 3, mt = job & 3;
      const unsigned short* arow = src + (size_t)(mt * 16 + lm) * srcStride;
      const float* bias = (mat == 0) ? P.b_kr : (mat == 1) ? P.b_kw
                        : (mat == 2) ? P.b_e  : P.b_a;
      int col = nt * 16 + lm;
      v8f acc = v8_splat(bias[col]);                   // bias folded into C init
      for (int kk = 0; kk < KT_H; kk++) {
        v16u a = load_a(arow + kk * 32, hl);
        v16u b = *(const v16u*)(P.WhF + (((size_t)(mat * 16 + kk) * 4 + nt) * 32 + lane) * 16);
        acc = wmma_bf16(a, b, acc);
      }
#pragma unroll
      for (int v = 0; v < 8; v++) {
        int row = mt * 16 + v + 8 * hl;
        float x = acc[v];
        if (mat == 2) x = sigm(x);                     // erase head -> sigmoid
        keyb[mat][row][col] = x;
      }
    }
  };

  // ======================= recurrent scan ==================================
  for (int t = 0; t < SEQ; t++) {
    // -- stream this step's embeddings into x_cat (bf16) --
    for (int idx = tid; idx < BSZ * IDIM; idx += THREADS) {
      int b = idx >> 8, k = idx & 255;
      xcat[b][k] = f2bf(P.embs[((size_t)t * BSZ + b) * IDIM + k]);
    }
    __syncthreads();

    // -- LSTM gate GEMM [64x832]x[832x2048] + fused elementwise cell --------
    unsigned short hreg[4][8];
    const int j = wave * 16 + lm;                      // hidden column this lane owns
    float bi_[4];
#pragma unroll
    for (int g = 0; g < 4; g++) bi_[g] = P.b_ih[g * CDIM + j] + P.b_hh[g * CDIM + j];
    for (int mt = 0; mt < 4; mt++) {
      v8f ai = v8_splat(bi_[0]), af = v8_splat(bi_[1]);
      v8f ag = v8_splat(bi_[2]), ao = v8_splat(bi_[3]);
      const unsigned short* arow = &xcat[mt * 16 + lm][0];
      for (int kk = 0; kk < KT_G; kk++) {
        v16u a = load_a(arow + kk * 32, hl);
        const unsigned short* wb = P.WgF + (size_t)kk * NT_G * 512;
        ai = wmma_bf16(a, *(const v16u*)(wb + ((size_t)(0 * 32 + wave)) * 512 + lane * 16), ai);
        af = wmma_bf16(a, *(const v16u*)(wb + ((size_t)(1 * 32 + wave)) * 512 + lane * 16), af);
        ag = wmma_bf16(a, *(const v16u*)(wb + ((size_t)(2 * 32 + wave)) * 512 + lane * 16), ag);
        ao = wmma_bf16(a, *(const v16u*)(wb + ((size_t)(3 * 32 + wave)) * 512 + lane * 16), ao);
      }
#pragma unroll
      for (int v = 0; v < 8; v++) {
        int b = mt * 16 + v + 8 * hl;                  // batch row (C/D layout)
        float cold = P.cW[b * CDIM + j];
        float cn = sigm(af[v]) * cold + sigm(ai[v]) * ftanh(ag[v]);
        float hn = sigm(ao[v]) * ftanh(cn);
        P.cW[b * CDIM + j] = cn;
        P.hW[b * CDIM + j] = hn;
        hreg[mt][v] = f2bf(hn);                        // defer LDS write (avoid RAW race)
      }
    }
    __syncthreads();                                   // everyone done reading old h
#pragma unroll
    for (int mt = 0; mt < 4; mt++)
#pragma unroll
      for (int v = 0; v < 8; v++)
        xcat[mt * 16 + v + 8 * hl][IDIM + MDIM + j] = hreg[mt][v];
    __syncthreads();

    // -- head projections for co = h (shared by hop 0 and hop 1) ------------
    heads_pass(&xcat[0][IDIM + MDIM], 840);
    __syncthreads();

    // -- hop 0: read result is dead code; only the write matters ------------
    scores_pass(keyb[1]); __syncthreads();             // kw
    softmax_pass();       __syncthreads();
    mem_update();         __syncthreads();

    // -- hop 1: read (feeds MLP), then write --------------------------------
    scores_pass(keyb[0]); __syncthreads();             // kr
    softmax_pass();       __syncthreads();
    read_pass(t, false);  __syncthreads();             // r1 -> xcat r-slot (bf16)
    scores_pass(keyb[1]); __syncthreads();
    softmax_pass();       __syncthreads();
    mem_update();         __syncthreads();

    // -- MLP: co2 = relu([h|r1] @ W_mlp + b); also emit os[t][:, :512] ------
    for (int q = 0; q < 4; q++) {
      int job = wave * 4 + q;                          // 128 tile jobs
      int mt = job >> 5, nt = job & 31;
      int row0 = mt * 16 + lm;
      int col = nt * 16 + lm;
      v8f acc = v8_splat(P.b_mlp[col]);                // bias folded into C init
      for (int kk = 0; kk < KT_MLP; kk++) {
        const unsigned short* ap = (kk < 16)
            ? &xcat[row0][IDIM + MDIM + kk * 32]       // co (=h) part, K<512
            : &xcat[row0][IDIM + (kk - 16) * 32];      // r1 part, K>=512
        v16u a = load_a(ap, hl);
        v16u b = *(const v16u*)(P.WmF + (((size_t)kk * 32 + nt) * 32 + lane) * 16);
        acc = wmma_bf16(a, b, acc);
      }
#pragma unroll
      for (int v = 0; v < 8; v++) {
        int row = mt * 16 + v + 8 * hl;
        float x = fmaxf(acc[v], 0.f);
        P.out[((size_t)t * BSZ + row) * (CDIM + MDIM) + col] = x;
        co2s[row][col] = f2bf(x);
      }
    }
    __syncthreads();

    // -- head projections for co = co2 (hop 2) ------------------------------
    heads_pass(&co2s[0][0], 520);
    __syncthreads();

    // -- hop 2: read (-> output + carry), then write ------------------------
    scores_pass(keyb[0]); __syncthreads();
    softmax_pass();       __syncthreads();
    read_pass(t, true);   __syncthreads();             // r2 -> out + xcat r-slot
    scores_pass(keyb[1]); __syncthreads();
    softmax_pass();       __syncthreads();
    mem_update();

    // -- capture hT/cT for sequences ending at this step --------------------
    for (int idx = tid; idx < BSZ * CDIM; idx += THREADS) {
      int b = idx >> 9;
      if (P.lens[b] - 1 == t) {
        P.out[OS_SZ + idx]              = P.hW[idx];
        P.out[OS_SZ + BSZ * CDIM + idx] = P.cW[idx];
      }
    }
    __syncthreads();
  }
}

// ---------------------------------------------------------------- launch
extern "C" void kernel_launch(void* const* d_in, const int* in_sizes, int n_in,
                              void* d_out, int out_size, void* d_ws, size_t ws_size,
                              hipStream_t stream) {
  const float* embs = (const float*)d_in[0];
  const int*   lens = (const int*)d_in[1];
  const float* Wih  = (const float*)d_in[2];
  const float* Whh  = (const float*)d_in[3];
  const float* b_ih = (const float*)d_in[4];
  const float* b_hh = (const float*)d_in[5];
  const float* h0   = (const float*)d_in[6];
  const float* c0   = (const float*)d_in[7];
  const float* r0   = (const float*)d_in[8];
  const float* mem0 = (const float*)d_in[9];
  const float* Wkr  = (const float*)d_in[10];
  const float* bkr  = (const float*)d_in[11];
  const float* Wkw  = (const float*)d_in[12];
  const float* bkw  = (const float*)d_in[13];
  const float* We   = (const float*)d_in[14];
  const float* be   = (const float*)d_in[15];
  const float* Wa   = (const float*)d_in[16];
  const float* ba   = (const float*)d_in[17];
  const float* Wm   = (const float*)d_in[18];
  const float* bm   = (const float*)d_in[19];

  char* ws = (char*)d_ws;
  size_t off = 0;
  auto carve = [&](size_t bytes) -> char* {
    char* p = ws + off; off += (bytes + 255) & ~(size_t)255; return p;
  };
  unsigned short* WgF = (unsigned short*)carve((size_t)(26 << 16) * 2);   // 3.25 MB
  unsigned short* WhF = (unsigned short*)carve((size_t)(4 << 15) * 2);    // 256 KB
  unsigned short* WmF = (unsigned short*)carve((size_t)(18 << 14) * 2);   // 576 KB
  float* memW = (float*)carve((size_t)BSZ * NSLOT * MDIM * 4);            // 2 MB
  float* hW   = (float*)carve((size_t)BSZ * CDIM * 4);
  float* cW   = (float*)carve((size_t)BSZ * CDIM * 4);

  prep_wg  <<<((26 << 16) + 255) / 256, 256, 0, stream>>>(Wih, Whh, WgF);
  prep_wh  <<<((4 << 15) + 255) / 256, 256, 0, stream>>>(Wkr, Wkw, We, Wa, WhF);
  prep_wmlp<<<((18 << 14) + 255) / 256, 256, 0, stream>>>(Wm, WmF);
  prep_state<<<(BSZ * NSLOT * MDIM + 255) / 256, 256, 0, stream>>>(mem0, h0, c0,
                                                                   memW, hW, cW);

  Params P;
  P.embs = embs; P.lens = lens; P.b_ih = b_ih; P.b_hh = b_hh;
  P.h0 = h0; P.c0 = c0; P.r0 = r0;
  P.b_kr = bkr; P.b_kw = bkw; P.b_e = be; P.b_a = ba; P.b_mlp = bm;
  P.WgF = WgF; P.WhF = WhF; P.WmF = WmF;
  P.memW = memW; P.hW = hW; P.cW = cW;
  P.out = (float*)d_out;

  mann_persistent<<<1, THREADS, 0, stream>>>(P);
}
